// CausalSelfAttention_18975165514359
// MI455X (gfx1250) — matrix operand order repaired and checked by reading, not compiled, not run on hardware
//
#include <hip/hip_runtime.h>

// MI455X (gfx1250) CDNA5 causal self-attention.
// - all matmuls via v_wmma_f32_16x16x32_f16 (wave32 WMMA, f32 accum)
// - f16 intermediates in d_ws: Q|K|V [B,H,S,D] + O [B,S,C]  (64 MB)
// - K/Q tiles in the attention kernel fetched by the Tensor Data Mover
//   (tensor_load_to_lds, double-buffered, s_wait_tensorcnt)
// - packed f32->f16 conversion (v_cvt_pk_rtz_f16_f32) for LDS staging

typedef _Float16 h16;
typedef __attribute__((ext_vector_type(2)))  __fp16   f16x2;   // cvt_pkrtz result type
typedef __attribute__((ext_vector_type(16))) _Float16 v16h;
typedef __attribute__((ext_vector_type(8)))  _Float16 v8h;
typedef __attribute__((ext_vector_type(8)))  float    v8f;
typedef __attribute__((ext_vector_type(4)))  unsigned int u32x4;
typedef __attribute__((ext_vector_type(8)))  int      i32x8;
typedef __attribute__((ext_vector_type(4)))  int      i32x4;

#define S_LEN 2048
#define C_DIM 1024
#define HEADS 16
#define HD    64
#define NEG_BIG (-3.0e38f)
// log2(10000)/32  (inv_freq = 2^(-(d&31)*log2(1e4)/32))
#define L2B_OVER_32 0.41524101186092f

#if defined(__has_builtin)
#if __has_builtin(__builtin_amdgcn_tensor_load_to_lds) && \
    __has_builtin(__builtin_amdgcn_s_wait_tensorcnt)
#define HAVE_TDM 1
#endif
#endif
#ifndef HAVE_TDM
#define HAVE_TDM 0
#endif

// packed f32 pair -> f16 pair store (v_cvt_pk_rtz_f16_f32 + ds_store_b32)
__device__ __forceinline__ void store_pk(h16* dst, float a, float b) {
  *(f16x2*)dst = __builtin_amdgcn_cvt_pkrtz(a, b);
}

// ---------------- WMMA fragment helpers (CDNA5 ISA 7.12.2 layouts) -------------

// A 16x32 f16: lane&15 = row M; lanes 0-15: K {0..7,16..23}, lanes 16-31: K {8..15,24..31}
__device__ __forceinline__ v16h frag_a(const h16* p, int lda, int lane) {
  const h16* row = p + (size_t)(lane & 15) * lda + ((lane >> 4) << 3);
  v8h lo = *(const v8h*)(row);
  v8h hi = *(const v8h*)(row + 16);
  return __builtin_shufflevector(lo, hi, 0,1,2,3,4,5,6,7,8,9,10,11,12,13,14,15);
}

// B 32x16 f16 from LDS stored [n][k]: lane&15 = col N; lanes 0-15: K {0..7,16..23}, lanes 16-31: K {8..15,24..31}
__device__ __forceinline__ v16h frag_b(const h16* p, int ldb, int lane) {
  const h16* row = p + (size_t)(lane & 15) * ldb + ((lane >> 4) << 3);
  v8h lo = *(const v8h*)(row);
  v8h hi = *(const v8h*)(row + 16);
  return __builtin_shufflevector(lo, hi, 0,1,2,3,4,5,6,7,8,9,10,11,12,13,14,15);
}

__device__ __forceinline__ v8f wmma_f16(v16h a, v16h b, v8f c) {
  return __builtin_amdgcn_wmma_f32_16x16x32_f16(false, a, false, b, (short)0, c,
                                                false, false);
}

#if HAVE_TDM
// Low 32 bits of a flat shared pointer == LDS byte offset (ISA 10.2 aperture map).
__device__ __forceinline__ unsigned lds_off(const void* p) {
  return (unsigned)(uintptr_t)p;
}
// 1-D contiguous f16 TDM load: nelems f16 from g -> LDS. D# per CDNA5 ISA.
__device__ __forceinline__ void tdm_load_f16(unsigned ldsoff, const h16* g,
                                             unsigned nelems) {
  unsigned long long ga = (unsigned long long)(uintptr_t)g;
  u32x4 g0;
  g0[0] = 1u;                                            // count=1 (valid), user mode
  g0[1] = ldsoff;                                        // lds_addr
  g0[2] = (unsigned)(ga & 0xffffffffu);                  // global_addr[31:0]
  g0[3] = (unsigned)((ga >> 32) & 0x01ffffffu) | (2u << 30);  // addr[56:32] | type=2
  i32x8 g1 = {0, 0, 0, 0, 0, 0, 0, 0};
  g1[0] = (int)(1u << 16);                               // data_size = 2 bytes
  g1[1] = (int)((nelems & 0xffffu) << 16);               // tensor_dim0[15:0]
  g1[2] = (int)(((nelems >> 16) & 0xffffu) | (1u << 16)); // dim0[31:16], tensor_dim1=1
  g1[3] = (int)((nelems & 0xffffu) << 16);               // tile_dim0 = nelems
  g1[5] = (int)nelems;                                   // tensor_dim0_stride
  i32x4 z4 = {0, 0, 0, 0};
  i32x8 z8 = {0, 0, 0, 0, 0, 0, 0, 0};
  __builtin_amdgcn_tensor_load_to_lds(g0, g1, z4, z4, z8, 0);
}
#endif

// ---------------- Kernel 1: qkv = x @ W_qkv + b, RoPE(q,k), scatter to [B,H,S,D] ----

__global__ __launch_bounds__(128) void qkv_rope_kernel(
    const float* __restrict__ x, const float* __restrict__ Wqkv,
    const float* __restrict__ bqkv,
    h16* __restrict__ Qb, h16* __restrict__ Kb, h16* __restrict__ Vb) {
  __shared__ h16 As[64][32];   // A tile, row-major
  __shared__ h16 Bs[64][32];   // B tile transposed: [n][k]

  const int m0 = blockIdx.x * 64;
  const int n0 = blockIdx.y * 64;       // 64-wide => exactly one head
  const int t = threadIdx.x;
  const int wave = t >> 5, lane = t & 31;

  v8f acc[4];
#pragma unroll
  for (int i = 0; i < 4; ++i) acc[i] = (v8f){0,0,0,0,0,0,0,0};

  for (int k0 = 0; k0 < C_DIM; k0 += 32) {
    __syncthreads();
    {
      // A: 64x32 f32 -> f16 LDS, packed pair converts
      int r = t >> 1, cg = (t & 1) * 16;
      const float* src = x + (size_t)(m0 + r) * C_DIM + k0 + cg;
#pragma unroll
      for (int i = 0; i < 16; i += 2)
        store_pk(&As[r][cg + i], src[i], src[i + 1]);
      // B: each thread owns a k-pair (contiguous in [n][k]) x 8 columns
      int kp = t & 15, ng = (t >> 4) * 8;
      const float* b0 = Wqkv + (size_t)(k0 + 2 * kp) * (3 * C_DIM) + n0 + ng;
      const float* b1 = b0 + 3 * C_DIM;
#pragma unroll
      for (int i = 0; i < 8; ++i)
        store_pk(&Bs[ng + i][2 * kp], b0[i], b1[i]);
    }
    __syncthreads();
    v16h af = frag_a(&As[wave * 16][0], 32, lane);
#pragma unroll
    for (int nb = 0; nb < 4; ++nb) {
      v16h bf = frag_b(&Bs[nb * 16][0], 32, lane);
      acc[nb] = wmma_f16(af, bf, acc[nb]);
    }
  }

  // Epilogue: bias + RoPE (+1/sqrt(D) pre-scale on Q) + scatter as f16.
  const int region = n0 >> 10;  // 0=q, 1=k, 2=v
  const int mrow_base = m0 + wave * 16 + ((lane >> 4) << 3);

#pragma unroll
  for (int nb = 0; nb < 4; ++nb) {
    const int n = n0 + nb * 16 + (lane & 15);
    const float bias = bqkv[n];
    const int nm = n & (C_DIM - 1);
    const int h = nm >> 6, d = nm & 63;
    const float inv_f = exp2f(-(float)(d & 31) * L2B_OVER_32);
#pragma unroll
    for (int j = 0; j < 8; ++j) {
      const int m = mrow_base + j;
      const int b = m >> 11, s = m & (S_LEN - 1);
      float v = acc[nb][j] + bias;
      const size_t dstoff = (((size_t)(b * HEADS + h)) * S_LEN + s) * HD + d;
      if (region == 2) {
        Vb[dstoff] = (h16)v;
      } else {
        // partner channel n^1 lives in lane^1 (C-layout: column = lane&15)
        float p = __shfl_xor(v, 1, 32);
        float sv, cv;
        __sincosf((float)s * inv_f, &sv, &cv);
        float o = v * cv + ((n & 1) ? p : -p) * sv;
        if (region == 0) {
          o *= 0.125f;  // 1/sqrt(64)
          Qb[dstoff] = (h16)o;
        } else {
          Kb[dstoff] = (h16)o;
        }
      }
    }
  }
}

// ---------------- Kernel 2: flash attention, one (b,h) x 64 query rows per block ----

__global__ __launch_bounds__(128) void attn_kernel(
    const h16* __restrict__ Qb, const h16* __restrict__ Kb,
    const h16* __restrict__ Vb, h16* __restrict__ Ob) {
  __shared__ h16 Qs[64][64];        // [qrow][d]
  __shared__ h16 Ks2[2][64][64];    // [key][d], double-buffered (TDM prefetch)
  __shared__ h16 Vt[64][64];        // [d][key]  == B-frag layout for P.V
  __shared__ h16 Ps[4][16][64];     // per-wave P scratch (A-frag source)

  const int bh = blockIdx.x;        // b*16+h
  const int qb = blockIdx.y;        // query block (64 rows)
  const int b = bh >> 4, h = bh & 15;
  const int t = threadIdx.x, wave = t >> 5, lane = t & 31;
  const int halfsel = lane >> 4;
  const size_t base = (size_t)bh * S_LEN * HD;

#if HAVE_TDM
  if (wave == 0) {   // TDM issues once per wave; EXEC ignored
    tdm_load_f16(lds_off(&Qs[0][0]), Qb + base + (size_t)qb * 64 * HD, 64 * HD);
    tdm_load_f16(lds_off(&Ks2[0][0][0]), Kb + base, 64 * HD);
  }
#else
  {  // manual Q tile copy
    int r = t >> 1, dg = (t & 1) * 32;
    const h16* src = Qb + base + (size_t)(qb * 64 + r) * HD + dg;
#pragma unroll
    for (int i = 0; i < 32; i += 8)
      *(v8h*)&Qs[r][dg + i] = *(const v8h*)(src + i);
  }
#endif

  float mrow[8], lrow[8];
  v8f oacc[4];
#pragma unroll
  for (int j = 0; j < 8; ++j) { mrow[j] = NEG_BIG; lrow[j] = 0.f; }
#pragma unroll
  for (int i = 0; i < 4; ++i) oacc[i] = (v8f){0,0,0,0,0,0,0,0};

  const int qidx_base = qb * 64 + wave * 16 + halfsel * 8;  // + j

  for (int kb = 0; kb <= qb; ++kb) {
    const int buf = kb & 1;
    __syncthreads();  // protect Vt (and Ks on the manual path)

    {  // V tile, transposed into Vt[d][key]; thread owns a key-pair -> packed b32 stores
      int kp = t & 31, dg2 = (t >> 5) * 16;
      const h16* v0 = Vb + base + (size_t)(kb * 64 + 2 * kp) * HD + dg2;
      const h16* v1 = v0 + HD;
      v8h a0 = *(const v8h*)(v0), a1 = *(const v8h*)(v0 + 8);
      v8h c0 = *(const v8h*)(v1), c1 = *(const v8h*)(v1 + 8);
#pragma unroll
      for (int i = 0; i < 8; ++i) {
        h16 p0[2] = {a0[i], c0[i]};
        *(f16x2*)&Vt[dg2 + i][2 * kp] = *(f16x2*)p0;
        h16 p1[2] = {a1[i], c1[i]};
        *(f16x2*)&Vt[dg2 + 8 + i][2 * kp] = *(f16x2*)p1;
      }
    }

#if HAVE_TDM
    if (wave == 0) {
      if (kb < qb) {  // prefetch next K tile into the other buffer
        tdm_load_f16(lds_off(&Ks2[buf ^ 1][0][0]),
                     Kb + base + (size_t)(kb + 1) * 64 * HD, 64 * HD);
        __builtin_amdgcn_s_wait_tensorcnt(1);  // current buffer complete
      } else {
        __builtin_amdgcn_s_wait_tensorcnt(0);
      }
    }
#else
    {  // manual K tile copy
      int r = t >> 1, dg = (t & 1) * 32;
      const h16* ks = Kb + base + (size_t)(kb * 64 + r) * HD + dg;
#pragma unroll
      for (int i = 0; i < 32; i += 8)
        *(v8h*)&Ks2[buf][r][dg + i] = *(const v8h*)(ks + i);
    }
#endif
    __syncthreads();

    // S = Q.K^T : 16x64 per wave
    v8f sacc[4];
#pragma unroll
    for (int i = 0; i < 4; ++i) sacc[i] = (v8f){0,0,0,0,0,0,0,0};
#pragma unroll
    for (int ks = 0; ks < 2; ++ks) {
      v16h aq = frag_a(&Qs[wave * 16][ks * 32], 64, lane);
#pragma unroll
      for (int nb = 0; nb < 4; ++nb) {
        v16h bk = frag_b(&Ks2[buf][nb * 16][ks * 32], 64, lane);
        sacc[nb] = wmma_f16(aq, bk, sacc[nb]);
      }
    }

    // causal mask + row max (16-lane half reductions; rows j / j+8 per half)
    float pmax[8];
#pragma unroll
    for (int j = 0; j < 8; ++j) {
      float mx = NEG_BIG;
#pragma unroll
      for (int nb = 0; nb < 4; ++nb) {
        int kk = kb * 64 + nb * 16 + (lane & 15);
        float s = sacc[nb][j];
        if (kk > qidx_base + j) s = NEG_BIG;
        sacc[nb][j] = s;
        mx = fmaxf(mx, s);
      }
      mx = fmaxf(mx, __shfl_xor(mx, 1, 16));
      mx = fmaxf(mx, __shfl_xor(mx, 2, 16));
      mx = fmaxf(mx, __shfl_xor(mx, 4, 16));
      mx = fmaxf(mx, __shfl_xor(mx, 8, 16));
      pmax[j] = mx;
    }

    // online softmax update; emit P (f16) to per-wave LDS in A-row layout
#pragma unroll
    for (int j = 0; j < 8; ++j) {
      float mnew = fmaxf(mrow[j], pmax[j]);
      float alpha = __expf(mrow[j] - mnew);
      mrow[j] = mnew;
      float psum = 0.f;
#pragma unroll
      for (int nb = 0; nb < 4; ++nb) {
        float p = __expf(sacc[nb][j] - mnew);
        psum += p;
        Ps[wave][halfsel * 8 + j][nb * 16 + (lane & 15)] = (h16)p;
      }
      psum += __shfl_xor(psum, 1, 16);
      psum += __shfl_xor(psum, 2, 16);
      psum += __shfl_xor(psum, 4, 16);
      psum += __shfl_xor(psum, 8, 16);
      lrow[j] = lrow[j] * alpha + psum;
#pragma unroll
      for (int db = 0; db < 4; ++db) oacc[db][j] = oacc[db][j] * alpha;
    }

    // same-wave LDS RAW: CDNA5 split dependency counter wait
    asm volatile("s_wait_dscnt 0x0" ::: "memory");

    // O += P.V : 16x64 per wave
#pragma unroll
    for (int ks = 0; ks < 2; ++ks) {
      v16h ap = frag_a(&Ps[wave][0][ks * 32], 64, lane);
#pragma unroll
      for (int db = 0; db < 4; ++db) {
        v16h bv = frag_b(&Vt[db * 16][ks * 32], 64, lane);
        oacc[db] = wmma_f16(ap, bv, oacc[db]);
      }
    }
  }

  // normalize and scatter to O[B,S,C] (f16), c = h*64 + d
#pragma unroll
  for (int j = 0; j < 8; ++j) {
    float inv = 1.f / lrow[j];
    int s = qidx_base + j;
    h16* dst = Ob + ((size_t)(b * S_LEN + s)) * C_DIM + h * HD;
#pragma unroll
    for (int db = 0; db < 4; ++db)
      dst[db * 16 + (lane & 15)] = (h16)(oacc[db][j] * inv);
  }
}

// ---------------- Kernel 3: out = O @ W_proj + b_proj (f32 output) ------------------

__global__ __launch_bounds__(128) void proj_kernel(
    const h16* __restrict__ A, const float* __restrict__ W,
    const float* __restrict__ bias, float* __restrict__ out) {
  __shared__ h16 As[64][32];
  __shared__ h16 Bs[64][32];  // [n][k]

  const int m0 = blockIdx.x * 64, n0 = blockIdx.y * 64;
  const int t = threadIdx.x, wave = t >> 5, lane = t & 31;

  v8f acc[4];
#pragma unroll
  for (int i = 0; i < 4; ++i) acc[i] = (v8f){0,0,0,0,0,0,0,0};

  for (int k0 = 0; k0 < C_DIM; k0 += 32) {
    __syncthreads();
    {
      int r = t >> 1, cg = (t & 1) * 16;
      const h16* src = A + (size_t)(m0 + r) * C_DIM + k0 + cg;
      *(v8h*)&As[r][cg]     = *(const v8h*)(src);
      *(v8h*)&As[r][cg + 8] = *(const v8h*)(src + 8);
      int kp = t & 15, ng = (t >> 4) * 8;
      const float* b0 = W + (size_t)(k0 + 2 * kp) * C_DIM + n0 + ng;
      const float* b1 = b0 + C_DIM;
#pragma unroll
      for (int i = 0; i < 8; ++i)
        store_pk(&Bs[ng + i][2 * kp], b0[i], b1[i]);
    }
    __syncthreads();
    v16h af = frag_a(&As[wave * 16][0], 32, lane);
#pragma unroll
    for (int nb = 0; nb < 4; ++nb) {
      v16h bf = frag_b(&Bs[nb * 16][0], 32, lane);
      acc[nb] = wmma_f16(af, bf, acc[nb]);
    }
  }

#pragma unroll
  for (int nb = 0; nb < 4; ++nb) {
    int n = n0 + nb * 16 + (lane & 15);
    float bv = bias[n];
#pragma unroll
    for (int j = 0; j < 8; ++j) {
      int m = m0 + wave * 16 + ((lane >> 4) << 3) + j;
      out[(size_t)m * C_DIM + n] = acc[nb][j] + bv;
    }
  }
}

// ---------------- host launch -------------------------------------------------------

extern "C" void kernel_launch(void* const* d_in, const int* in_sizes, int n_in,
                              void* d_out, int out_size, void* d_ws, size_t ws_size,
                              hipStream_t stream) {
  const float* x     = (const float*)d_in[0];
  const float* Wqkv  = (const float*)d_in[1];
  const float* bqkv  = (const float*)d_in[2];
  const float* Wproj = (const float*)d_in[3];
  const float* bproj = (const float*)d_in[4];
  float* out = (float*)d_out;

  const size_t per = (size_t)4 * HEADS * S_LEN * HD;  // 8,388,608 halfs = 16 MB
  h16* Qb = (h16*)d_ws;
  h16* Kb = Qb + per;
  h16* Vb = Kb + per;
  h16* Ob = Vb + per;  // [B,S,C] f16

  // M = B*S = 8192
  qkv_rope_kernel<<<dim3(128, 48), 128, 0, stream>>>(x, Wqkv, bqkv, Qb, Kb, Vb);
  attn_kernel<<<dim3(64, 32), 128, 0, stream>>>(Qb, Kb, Vb, Ob);
  proj_kernel<<<dim3(128, 16), 128, 0, stream>>>(Ob, Wproj, bproj, out);
}
